// InstanceEmbeddingLoss_22282290332167
// MI455X (gfx1250) — compile-verified, matrix operand order checked by reading, not run on hardware
//
#include <hip/hip_runtime.h>

typedef __attribute__((ext_vector_type(16))) _Float16 v16h;
typedef __attribute__((ext_vector_type(8)))  float    v8f;

// errors live in [0, 2]; monotone sortable key for non-negative floats is the
// raw bit pattern. bin = bits >> 13  -> 2^10 bins per octave, max bin 0x20000.
#define BIN_SHIFT 13
#define NBINS     (1u << 18)          // allocated per image (max used = 0x20000)
#define TOPBIN1   0x1FFFFu            // errors just below 2.0 (hot)
#define TOPBIN2   0x20000u            // exactly 2.0
#define NPIX      (512 * 512)
#define NCLS      48

// ---------------------------------------------------------------- init -----
__global__ void iel_init_kernel(unsigned long long* __restrict__ hist,
                                float* __restrict__ sums,
                                float* __restrict__ out) {
  size_t i = (size_t)blockIdx.x * blockDim.x + threadIdx.x;
  size_t stride = (size_t)gridDim.x * blockDim.x;
  size_t n = 4ull * NBINS;
  for (size_t k = i; k < n; k += stride) hist[k] = 0ull;
  if (i < 4 * NCLS * 3) sums[i] = 0.0f;
  if (i == 0) out[0] = 0.0f;
}

// ------------------------------------------------- per-class sums (exact) --
__global__ void iel_class_sums_kernel(const float* __restrict__ emb,
                                      const int* __restrict__ gt,
                                      float* __restrict__ sums) {
  __shared__ float ls[NCLS * 3];
  int t = threadIdx.x;
  if (t < NCLS * 3) ls[t] = 0.0f;
  __syncthreads();
  int img = blockIdx.x >> 6;      // 64 blocks per image (grid = 256)
  int blk = blockIdx.x & 63;
  const float* e0 = emb + (size_t)img * 2 * NPIX;
  const float* e1 = e0 + NPIX;
  const int*   g  = gt + (size_t)img * NPIX;
  for (int p = blk * 256 + t; p < NPIX; p += 64 * 256) {
    int l = g[p];
    if (l > 0) {
      atomicAdd(&ls[(l - 1) * 3 + 0], e0[p]);
      atomicAdd(&ls[(l - 1) * 3 + 1], e1[p]);
      atomicAdd(&ls[(l - 1) * 3 + 2], 1.0f);
    }
  }
  __syncthreads();
  if (t < NCLS * 3) {
    float v = ls[t];
    if (v != 0.0f) atomicAdd(&sums[img * NCLS * 3 + t], v);
  }
}

// ------------------------------------------------------- centers, |c|^2 ----
__global__ void iel_finalize_kernel(const float* __restrict__ sums,
                                    float* __restrict__ centers,
                                    float* __restrict__ gts) {
  int t = threadIdx.x;  // 1 block, 256 threads
  if (t < 4 * NCLS) {
    const float* s = sums + t * 3;
    float cnt = s[2];
    float inv = cnt > 0.0f ? 1.0f / cnt : 0.0f;
    float c0 = s[0] * inv, c1 = s[1] * inv;
    float* c = centers + t * 3;
    c[0] = c0; c[1] = c1; c[2] = c0 * c0 + c1 * c1;
  }
  if (t < 4) {
    float tot = 0.0f;
    for (int n = 0; n < NCLS; ++n) tot += sums[t * NCLS * 3 + n * 3 + 2];
    gts[t] = tot;
  }
}

// -------------------------- WMMA cross terms + error histogram (hot loop) --
// WMMA computes  (-2c).x + |c|^2  directly (A K0/K1 = -2c, K2 = |c|^2;
//                                           B K0/K1 = x,   K2 = 1.0)
// d^2 = wmma + |x|^2 (f32).  probs = 2^(-100 d^2)  since sigma = 0.1/sqrt(2ln2)
// err = lab ? 2-2p : 2p
__global__ void iel_errors_hist_kernel(const float* __restrict__ emb,
                                       const int* __restrict__ gt,
                                       const float* __restrict__ centers,
                                       unsigned long long* __restrict__ hist) {
  int tid  = threadIdx.x;
  int lane = tid & 31;
  int wave   = (int)((blockIdx.x * blockDim.x + tid) >> 5);
  int nwaves = (int)((gridDim.x * blockDim.x) >> 5);
  const int tilesPerImg = NPIX / 16;

  for (int img = 0; img < 4; ++img) {
    // A fragments (ISA 7.12.2): lanes 0-15 hold row M=lane, VGPR0={K0,K1},
    // VGPR1={K2,K3}.  K0=-2c0, K1=-2c1, K2=|c|^2.
    v16h afrag0 = {}, afrag1 = {}, afrag2 = {};
    if (lane < 16) {
      const float* cb = centers + (size_t)(img * NCLS) * 3;
      const float* c0p = cb + (0 * 16 + lane) * 3;
      const float* c1p = cb + (1 * 16 + lane) * 3;
      const float* c2p = cb + (2 * 16 + lane) * 3;
      afrag0[0] = (_Float16)(-2.0f * c0p[0]); afrag0[1] = (_Float16)(-2.0f * c0p[1]);
      afrag0[2] = (_Float16)(c0p[2]);
      afrag1[0] = (_Float16)(-2.0f * c1p[0]); afrag1[1] = (_Float16)(-2.0f * c1p[1]);
      afrag1[2] = (_Float16)(c1p[2]);
      afrag2[0] = (_Float16)(-2.0f * c2p[0]); afrag2[1] = (_Float16)(-2.0f * c2p[1]);
      afrag2[2] = (_Float16)(c2p[2]);
    }
    const float* e0 = emb + (size_t)img * 2 * NPIX;
    const float* e1 = e0 + NPIX;
    const int*   g  = gt + (size_t)img * NPIX;
    unsigned long long* hb = hist + (size_t)img * NBINS;

    // hot-bin aggregation (u32 pairs; <= ~100 elements/wave/image, no overflow)
    unsigned a0t = 0, a0p = 0, a1t = 0, a1p = 0, a2t = 0, a2p = 0;
    int mbase = (lane >> 4) << 3;                        // C rows: j or j+8

    for (int tile = wave; tile < tilesPerImg; tile += nwaves) {
      // All 32 lanes load pixel (lane&15): duplicated addresses coalesce to
      // one cacheline; avoids ds_bpermute + dscnt wait in the hot loop.
      int p = tile * 16 + (lane & 15);
      float x0 = e0[p];
      float x1 = e1[p];
      int   gtn = g[p];
      float xx2n = x0 * x0 + x1 * x1;
      int   gtrel = gtn - 1 - mbase;     // isPos <=> gtrel == b*16 + j (const)
      int pn = p + nwaves * 16;                          // next tile for wave
      __builtin_prefetch(e0 + pn, 0, 1);                 // global_prefetch_b8
      __builtin_prefetch(e1 + pn, 0, 1);

      // B fragment: lanes 0-15 hold column N=lane (its own pixel);
      // K0=x0, K1=x1, K2=1.0.
      v16h bfrag = {};
      if (lane < 16) {
        bfrag[0] = (_Float16)x0;
        bfrag[1] = (_Float16)x1;
        bfrag[2] = (_Float16)1.0f;
      }
      v8f czero = {};

      for (int b = 0; b < 3; ++b) {
        v16h af = (b == 0) ? afrag0 : (b == 1) ? afrag1 : afrag2;
        v8f d = __builtin_amdgcn_wmma_f32_16x16x32_f16(
            false, af, false, bfrag, (short)0, czero, false, false);
#pragma unroll
        for (int j = 0; j < 8; ++j) {
          float d2 = fmaxf(d[j] + xx2n, 0.0f);
          // 2 * 2^(-100 d^2) via a single v_exp_f32
          float e = __builtin_amdgcn_exp2f(fmaf(-100.0f, d2, 1.0f));
          unsigned isPos = (gtrel == b * 16 + j) ? 1u : 0u;  // inline-const cmp
          float err = isPos ? (2.0f - e) : e;
          unsigned bin = __float_as_uint(err) >> BIN_SHIFT;
          // branchless hot-bin accumulation (v_cmp + v_add), one branch left
          unsigned h0 = (bin == 0u)      ? 1u : 0u;
          unsigned h1 = (bin == TOPBIN1) ? 1u : 0u;
          unsigned h2 = (bin >= TOPBIN2) ? 1u : 0u;
          a0t += h0; a0p += h0 & isPos;
          a1t += h1; a1p += h1 & isPos;
          a2t += h2; a2p += h2 & isPos;
          if (bin - 1u < TOPBIN1 - 1u) {                 // 1 <= bin < TOPBIN1
            atomicAdd(&hb[bin], 1ull | ((unsigned long long)isPos << 32));
          }
        }
      }
    }
    if (a0t) atomicAdd(&hb[0],       (unsigned long long)a0t | ((unsigned long long)a0p << 32));
    if (a1t) atomicAdd(&hb[TOPBIN1], (unsigned long long)a1t | ((unsigned long long)a1p << 32));
    if (a2t) atomicAdd(&hb[TOPBIN2], (unsigned long long)a2t | ((unsigned long long)a2p << 32));
  }
}

// ----------------- descending scan over bins -> Lovász loss (tie-grouped) --
__global__ void iel_scan_loss_kernel(const unsigned long long* __restrict__ hist,
                                     const float* __restrict__ gts,
                                     float* __restrict__ out) {
  __shared__ unsigned long long sc[256];
  __shared__ float lr[256];
  int img = blockIdx.x;
  int t = threadIdx.x;
  float gtsv = fmaxf(gts[img], 1.0f);
  const unsigned long long* h = hist + (size_t)img * NBINS;
  unsigned long long carry = 0;
  float loss = 0.0f;

  for (int chunk = 0; chunk <= (int)TOPBIN2; chunk += 256) {
    int idx = (int)TOPBIN2 - (chunk + t);                // descending bins
    unsigned long long hv = (idx >= 0) ? h[idx] : 0ull;
    sc[t] = hv;
    __syncthreads();
    for (int off = 1; off < 256; off <<= 1) {            // inclusive scan
      unsigned long long v = (t >= off) ? sc[t - off] : 0ull;
      __syncthreads();
      sc[t] += v;
      __syncthreads();
    }
    unsigned long long inc = sc[t] + carry;              // packed (tot|pos<<32)
    unsigned long long bef = inc - hv;
    unsigned tot = (unsigned)hv;
    if (tot && idx >= 0) {
      float tb = (float)(unsigned)bef,  pb = (float)(unsigned)(bef >> 32);
      float ta = (float)(unsigned)inc,  pa = (float)(unsigned)(inc >> 32);
      float Jb = 1.0f - (gtsv - pb) / fmaxf(gtsv + tb - pb, 1.0f);
      float Ja = 1.0f - (gtsv - pa) / fmaxf(gtsv + ta - pa, 1.0f);
      unsigned mid = (idx != 0 && idx != (int)TOPBIN2) ? (1u << (BIN_SHIFT - 1)) : 0u;
      float v = __uint_as_float(((unsigned)idx << BIN_SHIFT) | mid);
      loss += v * (Ja - Jb);                             // tie-group: v * dJ
    }
    __syncthreads();
    carry += sc[255];
    __syncthreads();
  }
  lr[t] = loss;
  __syncthreads();
  for (int off = 128; off > 0; off >>= 1) {
    if (t < off) lr[t] += lr[t + off];
    __syncthreads();
  }
  if (t == 0) atomicAdd(out, lr[0] * 0.25f);             // mean over 4 images
}

// ---------------------------------------------------------------------------
extern "C" void kernel_launch(void* const* d_in, const int* in_sizes, int n_in,
                              void* d_out, int out_size, void* d_ws, size_t ws_size,
                              hipStream_t stream) {
  const float* emb = (const float*)d_in[0];   // [4,2,512,512] f32
  const int*   gt  = (const int*)d_in[1];     // [4,1,512,512] i32
  float* out = (float*)d_out;                 // scalar f32

  // workspace: hist[4][NBINS] u64 (8.39 MB) | sums f32[4][48][3]
  //            | centers f32[4][48][3] (c0,c1,|c|^2) | gts f32[4]
  unsigned long long* hist = (unsigned long long*)d_ws;
  float* sums    = (float*)(hist + 4ull * NBINS);
  float* centers = sums + 4 * NCLS * 3;
  float* gts     = centers + 4 * NCLS * 3;

  iel_init_kernel<<<1024, 256, 0, stream>>>(hist, sums, out);
  iel_class_sums_kernel<<<256, 256, 0, stream>>>(emb, gt, sums);
  iel_finalize_kernel<<<1, 256, 0, stream>>>(sums, centers, gts);
  iel_errors_hist_kernel<<<512, 256, 0, stream>>>(emb, gt, centers, hist);
  iel_scan_loss_kernel<<<4, 256, 0, stream>>>(hist, gts, out);
}
